// MistralAttention_44246753083857
// MI455X (gfx1250) — matrix operand order repaired and checked by reading, not compiled, not run on hardware
//
#include <hip/hip_runtime.h>
#include <hip/hip_bf16.h>

// ---------------- CDNA5 WMMA types ----------------
typedef _Float16 h8  __attribute__((ext_vector_type(8)));
typedef _Float16 h16 __attribute__((ext_vector_type(16)));
typedef float    f8  __attribute__((ext_vector_type(8)));

// ---------------- Problem constants ----------------
#define BB   2
#define SS   2048
#define DD   128
#define HH   32
#define KVH  8
#define HD   128
#define WW   4096
#define MB   4
#define NQ   (HH*HD)       // 4096
#define NKV  (KVH*HD)      // 1024
#define MM   (BB*SS)       // 4096

// workspace byte offsets (fp16 staging)
#define OFF_QH    ((size_t)0)                         // 4096*4096 h = 32MB
#define OFF_ATTNH ((size_t)33554432)                  // 32MB
#define OFF_KRAW  ((size_t)67108864)                  // 4096*1024 h = 8MB
#define OFF_VRAW  ((size_t)75497472)                  // 8MB
#define OFF_XH    ((size_t)83886080)                  // 4096*128 h = 1MB
#define OFF_WQH   ((size_t)84934656)                  // 1MB
#define OFF_WKH   ((size_t)85983232)                  // 256KB
#define OFF_WVH   ((size_t)86245376)                  // 256KB
#define OFF_WOH   ((size_t)86507520)                  // 1MB

// d_out layout: out fp32 (B*S*D) | cache_k fp16 | cache_v fp16
#define OUT_F32_ELEMS ((size_t)BB*SS*DD)              // 524288
#define CACHE_ELEMS   ((size_t)MB*WW*KVH*HD)          // 16777216

static __device__ __forceinline__ f8 wmma_f16(h16 a, h16 b, f8 c) {
  return __builtin_amdgcn_wmma_f32_16x16x32_f16(false, a, false, b, (short)0, c,
                                                false, false);
}

// ---------------- fp32 -> fp16 convert ----------------
__global__ __launch_bounds__(256) void k_cvt_f32_to_f16(const float* __restrict__ src,
                                                        _Float16* __restrict__ dst,
                                                        unsigned n) {
  unsigned i = blockIdx.x * 256u + threadIdx.x;
  if (i < n) dst[i] = (_Float16)src[i];
}

// ---------------- copy fp16 (cache passthrough) ----------------
__global__ __launch_bounds__(256) void k_copy_h(const _Float16* __restrict__ src,
                                                _Float16* __restrict__ dst,
                                                unsigned n) {
  unsigned i = blockIdx.x * 256u + threadIdx.x;
  if (i < n) dst[i] = src[i];
}

// ---------------- generic WMMA GEMM: C(MxN) = A(MxK,f16) * B(KxN,f16) ------
// one wave per 16x16 output tile; K consumed in chunks of 32
__global__ __launch_bounds__(32) void k_gemm_wmma(const _Float16* __restrict__ A,
                                                  const _Float16* __restrict__ Bw,
                                                  float* __restrict__ Cf,
                                                  _Float16* __restrict__ Ch,
                                                  int N, int K,
                                                  int lda, int ldb, int ldc) {
  const int tiles_n = N >> 4;
  const int tn = blockIdx.x % tiles_n;
  const int tm = blockIdx.x / tiles_n;
  const int lane = threadIdx.x & 31;
  const int l16 = lane & 15;
  const int hf  = lane >> 4;
  const int m0 = tm << 4, n0 = tn << 4;

  f8 c = {};
  const _Float16* arow = A + (size_t)(m0 + l16) * lda;
  for (int kc = 0; kc < K; kc += 32) {
    // A-matrix: lane m=l16 holds two contiguous 8-half K chunks
    h8 a0 = *(const h8*)(arow + kc + hf * 8);
    h8 a1 = *(const h8*)(arow + kc + 16 + hf * 8);
    h16 a;
#pragma unroll
    for (int i = 0; i < 8; ++i) { a[i] = a0[i]; a[8 + i] = a1[i]; }
    // B-matrix: lane holds column N=n0+l16, K in the same order as A
    const int kb = kc + l16 + hf * 16;
    h16 b;
#pragma unroll
    for (int i = 0; i < 8; ++i) {
      b[i]     = Bw[(size_t)(kc + hf * 8 + i) * ldb + n0 + l16];
      b[8 + i] = Bw[(size_t)(kc + 16 + hf * 8 + i) * ldb + n0 + l16];
    }
    __builtin_prefetch(Bw + (size_t)(kb + 32) * ldb + n0, 0, 1);
    c = wmma_f16(a, b, c);
  }
#pragma unroll
  for (int v = 0; v < 8; ++v) {
    const int row = m0 + v + 8 * hf;
    const int col = n0 + l16;
    if (Cf) Cf[(size_t)row * ldc + col] = c[v];
    else    Ch[(size_t)row * ldc + col] = (_Float16)c[v];
  }
}

// ---------------- rotary (GPT-J interleaved) + cache scatter ----------------
__global__ __launch_bounds__(256) void k_rotary_scatter(_Float16* __restrict__ qh,
                                                        const _Float16* __restrict__ kraw,
                                                        const _Float16* __restrict__ vraw,
                                                        const float* __restrict__ freqs,
                                                        const int* __restrict__ positions,
                                                        _Float16* __restrict__ cacheK,
                                                        _Float16* __restrict__ cacheV) {
  const size_t Tq = (size_t)BB * SS * HH * (HD / 2);
  const size_t Tk = (size_t)BB * SS * KVH * (HD / 2);
  const size_t Tv = (size_t)BB * SS * KVH * HD;
  size_t i = (size_t)blockIdx.x * 256u + threadIdx.x;
  if (i < Tq) {
    int p = i % (HD / 2); size_t r = i / (HD / 2);
    int h = r % HH; r /= HH;
    int s = r % SS; int b = (int)(r / SS);
    float cs = freqs[(size_t)s * HD + 2 * p];
    float sn = freqs[(size_t)s * HD + 2 * p + 1];
    size_t base = ((((size_t)b * SS + s) * HH + h) * HD) + 2 * p;
    float x0 = (float)qh[base], x1 = (float)qh[base + 1];
    qh[base]     = (_Float16)(x0 * cs - x1 * sn);
    qh[base + 1] = (_Float16)(x0 * sn + x1 * cs);
  } else if (i < Tq + Tk) {
    size_t j = i - Tq;
    int p = j % (HD / 2); size_t r = j / (HD / 2);
    int kh = r % KVH; r /= KVH;
    int s = r % SS; int b = (int)(r / SS);
    float cs = freqs[(size_t)s * HD + 2 * p];
    float sn = freqs[(size_t)s * HD + 2 * p + 1];
    size_t src = ((((size_t)b * SS + s) * KVH + kh) * HD) + 2 * p;
    float x0 = (float)kraw[src], x1 = (float)kraw[src + 1];
    int pos = positions[s] & (WW - 1);
    size_t dst = ((((size_t)b * WW + pos) * KVH + kh) * HD) + 2 * p;
    cacheK[dst]     = (_Float16)(x0 * cs - x1 * sn);
    cacheK[dst + 1] = (_Float16)(x0 * sn + x1 * cs);
  } else if (i < Tq + Tk + Tv) {
    size_t j = i - Tq - Tk;
    int d = j % HD; size_t r = j / HD;
    int kh = r % KVH; r /= KVH;
    int s = r % SS; int b = (int)(r / SS);
    int pos = positions[s] & (WW - 1);
    size_t src = ((((size_t)b * SS + s) * KVH + kh) * HD) + d;
    size_t dst = ((((size_t)b * WW + pos) * KVH + kh) * HD) + d;
    cacheV[dst] = vraw[src];
  }
}

// ---------------- attention: one wave per (b, h, 16-row q block) ------------
// LDS: fp32 scores [16][SS] | fp16 probs [16][SS] | fp32 inv_sum[16]
__global__ __launch_bounds__(32) void k_attn(const _Float16* __restrict__ qh,
                                             const _Float16* __restrict__ cacheK,
                                             const _Float16* __restrict__ cacheV,
                                             const float* __restrict__ mask,
                                             _Float16* __restrict__ attnh) {
  extern __shared__ char smem[];
  float*    sScore = (float*)smem;                                   // 16*SS*4
  _Float16* sProb  = (_Float16*)(smem + (size_t)16 * SS * 4);        // 16*SS*2
  float*    sInv   = (float*)(smem + (size_t)16 * SS * 4 + (size_t)16 * SS * 2);

  const int QT = SS / 16;
  const int qt = blockIdx.x % QT;
  const int h  = (blockIdx.x / QT) % HH;
  const int b  = blockIdx.x / (QT * HH);
  const int kvh = h >> 2;                 // GQA: 32 q heads -> 8 kv heads
  const int lane = threadIdx.x & 31;
  const int l16 = lane & 15, hf = lane >> 4;
  const int q0 = qt * 16;
  const float scale = 0.088388347648318447f;   // 128^-0.5

  // Q in B-matrix layout (reused for every key tile): lane holds q-column l16,
  // 16 hd values in the same K order as A (gather, done once).
  h16 qB[4];
#pragma unroll
  for (int kc = 0; kc < 4; ++kc) {
#pragma unroll
    for (int n = 0; n < 16; ++n) {
      const int hd = kc * 32 + (n < 8 ? hf * 8 + n : 16 + hf * 8 + (n - 8));
      qB[kc][n] = qh[((((size_t)b * SS + q0 + l16) * HH + h) * HD) + hd];
    }
  }

  const _Float16* ckb = cacheK + (size_t)b * WW * KVH * HD + (size_t)kvh * HD;
  const _Float16* cvb = cacheV + (size_t)b * WW * KVH * HD + (size_t)kvh * HD;

  const int nkt = qt + 1;                 // causal: keys 0 .. q0+15
  for (int kt = 0; kt < nkt; ++kt) {
    const int key0 = kt * 16;
    const _Float16* krow = ckb + (size_t)(key0 + l16) * (KVH * HD);
    f8 sc = {};
#pragma unroll
    for (int kc = 0; kc < 4; ++kc) {
      h8 a0 = *(const h8*)(krow + kc * 32 + hf * 8);
      h8 a1 = *(const h8*)(krow + kc * 32 + 16 + hf * 8);
      h16 a;
#pragma unroll
      for (int i = 0; i < 8; ++i) { a[i] = a0[i]; a[8 + i] = a1[i]; }
      sc = wmma_f16(a, qB[kc], sc);       // scores^T tile: row=key, col=q
    }
#pragma unroll
    for (int v = 0; v < 8; ++v) {
      const int key = key0 + v + 8 * hf;
      const int q   = q0 + l16;
      const float val = sc[v] * scale + mask[(size_t)q * SS + key];
      sScore[(size_t)l16 * SS + key] = val;
    }
  }
  __syncthreads();

  const int klen   = nkt * 16;
  const int klen32 = ((nkt + 1) >> 1) << 5;   // pad key count to multiple of 32
  if (lane < 16) {
    float* row = sScore + (size_t)lane * SS;
    float m = -__builtin_inff();
    for (int j = 0; j < klen; ++j) m = fmaxf(m, row[j]);
    float sum = 0.f;
    _Float16* prow = sProb + (size_t)lane * SS;
    for (int j = 0; j < klen; ++j) {
      float e = __expf(row[j] - m);
      sum += e;
      prow[j] = (_Float16)e;
    }
    for (int j = klen; j < klen32; ++j) prow[j] = (_Float16)0.f;
    sInv[lane] = 1.0f / sum;
  }
  __syncthreads();

  f8 o[8];
#pragma unroll
  for (int nt = 0; nt < 8; ++nt) o[nt] = (f8){};
  for (int kk = 0; kk < klen32; kk += 32) {
    const _Float16* prow = sProb + (size_t)l16 * SS + kk;
    h8 a0 = *(const h8*)(prow + hf * 8);
    h8 a1 = *(const h8*)(prow + 16 + hf * 8);
    h16 a;
#pragma unroll
    for (int i = 0; i < 8; ++i) { a[i] = a0[i]; a[8 + i] = a1[i]; }
    const _Float16* vcol = cvb + (size_t)kk * (KVH * HD) + l16;
#pragma unroll
    for (int nt = 0; nt < 8; ++nt) {
      h16 bv;
#pragma unroll
      for (int i = 0; i < 8; ++i) {
        bv[i]     = vcol[(size_t)(hf * 8 + i) * (KVH * HD) + nt * 16];
        bv[8 + i] = vcol[(size_t)(16 + hf * 8 + i) * (KVH * HD) + nt * 16];
      }
      o[nt] = wmma_f16(a, bv, o[nt]);
    }
  }
#pragma unroll
  for (int nt = 0; nt < 8; ++nt) {
#pragma unroll
    for (int v = 0; v < 8; ++v) {
      const int row = v + 8 * hf;
      const float val = o[nt][v] * sInv[row];
      attnh[((((size_t)b * SS + q0 + row) * HH + h) * HD) + nt * 16 + l16] =
          (_Float16)val;
    }
  }
}

extern "C" void kernel_launch(void* const* d_in, const int* in_sizes, int n_in,
                              void* d_out, int out_size, void* d_ws, size_t ws_size,
                              hipStream_t stream) {
  const float* x     = (const float*)d_in[0];
  const float* wq    = (const float*)d_in[1];
  const float* wk    = (const float*)d_in[2];
  const float* wv    = (const float*)d_in[3];
  const float* wo    = (const float*)d_in[4];
  const float* freqs = (const float*)d_in[5];
  const int*   pos   = (const int*)d_in[6];
  const float* mask  = (const float*)d_in[7];
  const _Float16* cin_k = (const _Float16*)d_in[8];
  const _Float16* cin_v = (const _Float16*)d_in[9];

  char* ws = (char*)d_ws;
  _Float16* qh    = (_Float16*)(ws + OFF_QH);
  _Float16* attnh = (_Float16*)(ws + OFF_ATTNH);
  _Float16* kraw  = (_Float16*)(ws + OFF_KRAW);
  _Float16* vraw  = (_Float16*)(ws + OFF_VRAW);
  _Float16* xh    = (_Float16*)(ws + OFF_XH);
  _Float16* wqh   = (_Float16*)(ws + OFF_WQH);
  _Float16* wkh   = (_Float16*)(ws + OFF_WKH);
  _Float16* wvh   = (_Float16*)(ws + OFF_WVH);
  _Float16* woh   = (_Float16*)(ws + OFF_WOH);

  float* out = (float*)d_out;
  _Float16* cacheK = (_Float16*)((char*)d_out + OUT_F32_ELEMS * sizeof(float));
  _Float16* cacheV = cacheK + CACHE_ELEMS;

  // 1) fp32 -> fp16 staging
  auto cvt = [&](const float* s, _Float16* d, unsigned n) {
    k_cvt_f32_to_f16<<<(n + 255) / 256, 256, 0, stream>>>(s, d, n);
  };
  cvt(x,  xh,  MM * DD);
  cvt(wq, wqh, DD * NQ);
  cvt(wk, wkh, DD * NKV);
  cvt(wv, wvh, DD * NKV);
  cvt(wo, woh, NQ * DD);

  // 2) cache passthrough (entries not scattered keep their input values)
  k_copy_h<<<(unsigned)((CACHE_ELEMS + 255) / 256), 256, 0, stream>>>(cin_k, cacheK, (unsigned)CACHE_ELEMS);
  k_copy_h<<<(unsigned)((CACHE_ELEMS + 255) / 256), 256, 0, stream>>>(cin_v, cacheV, (unsigned)CACHE_ELEMS);

  // 3) QKV projections via WMMA
  k_gemm_wmma<<<(MM / 16) * (NQ / 16),  32, 0, stream>>>(xh, wqh, nullptr, qh,   NQ,  DD, DD, NQ,  NQ);
  k_gemm_wmma<<<(MM / 16) * (NKV / 16), 32, 0, stream>>>(xh, wkh, nullptr, kraw, NKV, DD, DD, NKV, NKV);
  k_gemm_wmma<<<(MM / 16) * (NKV / 16), 32, 0, stream>>>(xh, wvh, nullptr, vraw, NKV, DD, DD, NKV, NKV);

  // 4) rotary on q/k + scatter k/v into caches
  {
    const size_t total = (size_t)BB * SS * HH * (HD / 2)
                       + (size_t)BB * SS * KVH * (HD / 2)
                       + (size_t)BB * SS * KVH * HD;
    k_rotary_scatter<<<(unsigned)((total + 255) / 256), 256, 0, stream>>>(
        qh, kraw, vraw, freqs, pos, cacheK, cacheV);
  }

  // 5) causal GQA attention (WMMA Q·K^T and P·V, softmax in LDS)
  {
    const size_t lds = (size_t)16 * SS * 4 + (size_t)16 * SS * 2 + 64;
    k_attn<<<BB * HH * (SS / 16), 32, lds, stream>>>(qh, cacheK, cacheV, mask, attnh);
  }

  // 6) output projection -> fp32 out
  k_gemm_wmma<<<(MM / 16) * (DD / 16), 32, 0, stream>>>(attnh, woh, out, nullptr, DD, NQ, NQ, DD, DD);
}
